// InstAtt_75926431858915
// MI455X (gfx1250) — compile-verified, matrix operand order checked by reading, not run on hardware
//
#include <hip/hip_runtime.h>
#include <math.h>

typedef __bf16 bf16_t;
typedef __attribute__((ext_vector_type(16))) __bf16 v16bf;
typedef __attribute__((ext_vector_type(8)))  float   v8f;

#define WMMA_BF16(a, b, c) \
  __builtin_amdgcn_wmma_f32_16x16x32_bf16(false, (a), false, (b), (short)0, (c), false, false)

// ---------------------------------------------------------------------------
// bf16 weight staging layout in d_ws (element offsets):
//   Wih_att [256][128], Whh_att [256][64], Wih_lang [256][128],
//   Whh_lang [256][64], fc_w padded [2016][64]
// ---------------------------------------------------------------------------
__global__ void cvt_weights(const float* __restrict__ wia, const float* __restrict__ wha,
                            const float* __restrict__ wil, const float* __restrict__ whl,
                            const float* __restrict__ fcw, bf16_t* __restrict__ ws) {
  const int n0 = 256 * 128;
  const int n1 = n0 + 256 * 64;
  const int n2 = n1 + 256 * 128;
  const int n3 = n2 + 256 * 64;
  const int n4 = n3 + 2016 * 64;
  int i = blockIdx.x * blockDim.x + threadIdx.x;
  if (i < n0)      ws[i] = (__bf16)wia[i];
  else if (i < n1) ws[i] = (__bf16)wha[i - n0];
  else if (i < n2) ws[i] = (__bf16)wil[i - n1];
  else if (i < n3) ws[i] = (__bf16)whl[i - n2];
  else if (i < n4) {
    int j = i - n3;
    int row = j >> 6;                       // /64
    ws[i] = (row < 2001) ? (__bf16)fcw[j] : (__bf16)0.0f;
  }
}

// ----- WMMA fragment builders --------------------------------------------
// A fragment (16x32, bf16) from fp32 LDS tile, row-major, leading dim ld.
// Lane L<16: row M=L, K = k0+{0..7, 16..23}; Lane L>=16: row M=L-16,
// K = k0+{8..15, 24..31}  -> base = k0 + 8*hi, second group at +16.
__device__ inline v16bf a_frag_lds(const float* src, int ld, int k0, int lane) {
  const int m = lane & 15, hi = lane >> 4;
  const float* r = src + m * ld + k0 + 8 * hi;
  v16bf a;
#pragma unroll
  for (int e = 0; e < 8; ++e) { a[e] = (__bf16)r[e]; a[8 + e] = (__bf16)r[16 + e]; }
  return a;
}

// B fragment (32x16): W stored row-major [N][K]; W row n == B column n.
__device__ inline v16bf b_frag_g(const bf16_t* W, int ld, int n0, int k0, int lane) {
  const int n = n0 + (lane & 15), hi = lane >> 4;
  const bf16_t* r = W + n * ld + k0 + 8 * hi;
  v16bf b;
#pragma unroll
  for (int e = 0; e < 8; ++e) { b[e] = r[e]; b[8 + e] = r[16 + e]; }
  return b;
}

__device__ inline float sigf(float x) { return 1.0f / (1.0f + expf(-x)); }

// gates[16][256] = xin[16][128] @ Wih^T + hin[16][64] @ Whh^T + bih + bhh
// wave w computes N-tiles {w, w+8}. Uses 6 WMMAs per tile.
__device__ inline void lstm_gates(const float* xin, const float* hin,
                                  const bf16_t* Wih, const bf16_t* Whh,
                                  const float* bih, const float* bhh,
                                  float* gatesL, int wave, int lane) {
#pragma unroll
  for (int tt = 0; tt < 2; ++tt) {
    const int n0 = (wave + tt * 8) * 16;
    v8f acc = {};
#pragma unroll
    for (int k0 = 0; k0 < 128; k0 += 32)
      acc = WMMA_BF16(a_frag_lds(xin, 128, k0, lane), b_frag_g(Wih, 128, n0, k0, lane), acc);
#pragma unroll
    for (int k0 = 0; k0 < 64; k0 += 32)
      acc = WMMA_BF16(a_frag_lds(hin, 64, k0, lane), b_frag_g(Whh, 64, n0, k0, lane), acc);
    const int n = lane & 15, hi = lane >> 4;
    const float bias = bih[n0 + n] + bhh[n0 + n];
#pragma unroll
    for (int r = 0; r < 8; ++r)
      gatesL[(r + 8 * hi) * 256 + n0 + n] = acc[r] + bias;
  }
}

// ---------------------------------------------------------------------------
// Persistent kernel: each block owns 16 batch rows for all 47 steps.
// 256 threads = 8 waves.
// ---------------------------------------------------------------------------
__global__ __launch_bounds__(256) void instatt_main(
    const int* __restrict__ inst, const float* __restrict__ prevg,
    const float* __restrict__ finalg, const float* __restrict__ embed,
    const bf16_t* __restrict__ ws,
    const float* __restrict__ bia, const float* __restrict__ bha,
    const float* __restrict__ bil, const float* __restrict__ bhl,
    const float* __restrict__ fcb,
    const float* __restrict__ a1w1, const float* __restrict__ a1b1,
    const float* __restrict__ a1w2, const float* __restrict__ a1b2,
    const float* __restrict__ a2w1, const float* __restrict__ a2b1,
    const float* __restrict__ a2w2, const float* __restrict__ a2b2,
    const float* __restrict__ tw1, const float* __restrict__ tb1,
    const float* __restrict__ tw2, const float* __restrict__ tb2,
    float* __restrict__ out) {
  __shared__ float xbuf[16 * 128];
  __shared__ float gatesL[16 * 256];
  __shared__ float hatt[16 * 64], catt[16 * 64], hlang[16 * 64], clang[16 * 64];
  __shared__ float prevc[16 * 100], diffc[16 * 100];
  __shared__ float hproj1[16 * 32], hproj2[16 * 32];
  __shared__ float sc1[16 * 25], sc2[16 * 25];
  __shared__ float attv[16 * 8];
  __shared__ float ttmp[16 * 32];
  __shared__ float red[8 * 16];
  __shared__ float rowmax[16], rowlz[16];

  const int tid = threadIdx.x, lane = tid & 31, wave = tid >> 5;
  const int b0 = blockIdx.x * 16;

  const bf16_t* Wia = ws;
  const bf16_t* Wha = Wia + 256 * 128;
  const bf16_t* Wil = Wha + 256 * 64;
  const bf16_t* Whl = Wil + 256 * 128;
  const bf16_t* Fcw = Whl + 256 * 64;

  // ---- init state + canvases -------------------------------------------
  for (int i = tid; i < 16 * 64; i += 256) { hatt[i] = 0.f; catt[i] = 0.f; hlang[i] = 0.f; clang[i] = 0.f; }
  for (int i = tid; i < 16 * 100; i += 256) prevc[i] = prevg[b0 * 100 + i];
  __syncthreads();
  for (int i = tid; i < 16 * 25; i += 256) {
    const float* p = &prevc[i * 4];
    const bool msk = (p[0] + p[1] + p[2] + p[3]) > 0.f;
#pragma unroll
    for (int d = 0; d < 4; ++d)
      diffc[i * 4 + d] = msk ? -1.0f : finalg[b0 * 100 + i * 4 + d];
  }
  __syncthreads();

  for (int t = 0; t < 47; ++t) {
    // ---- x_att = [h_lang, embed[inst[:,t]]] -----------------------------
    for (int i = tid; i < 16 * 64; i += 256) {
      const int b = i >> 6, k = i & 63;
      xbuf[b * 128 + k] = hlang[i];
      const int iv = inst[(b0 + b) * 48 + t];
      xbuf[b * 128 + 64 + k] = embed[iv * 64 + k];
    }
    __syncthreads();
    lstm_gates(xbuf, hatt, Wia, Wha, bia, bha, gatesL, wave, lane);
    __syncthreads();
    for (int i = tid; i < 16 * 64; i += 256) {   // att-LSTM pointwise
      const int b = i >> 6, k = i & 63;
      const float* g = &gatesL[b * 256];
      const float ig = sigf(g[k]), fg = sigf(g[64 + k]);
      const float gg = tanhf(g[128 + k]), og = sigf(g[192 + k]);
      const float c2 = fg * catt[i] + ig * gg;
      catt[i] = c2;
      hatt[i] = og * tanhf(c2);
    }
    __syncthreads();

    // ---- attention: shared h-projection for both attends ----------------
    for (int i = tid; i < 16 * 32; i += 256) {
      const int b = i >> 5, j = i & 31;
      float s1 = a1b1[j], s2 = a2b1[j];
      const float* h = &hatt[b * 64];
      const float* w1 = &a1w1[j * 68];
      const float* w2 = &a2w1[j * 68];
      for (int k = 0; k < 64; ++k) { const float hv = h[k]; s1 += hv * w1[k]; s2 += hv * w2[k]; }
      hproj1[i] = s1; hproj2[i] = s2;
    }
    __syncthreads();
    for (int i = tid; i < 400; i += 256) {       // scores over 25 objects
      const int b = i / 25;
      float s1 = a1b2[0], s2 = a2b2[0];
      const float* m1 = &diffc[i * 4];
      const float* m2 = &prevc[i * 4];
      for (int j = 0; j < 32; ++j) {
        float u1 = hproj1[b * 32 + j] + m1[0] * a1w1[j * 68 + 64] + m1[1] * a1w1[j * 68 + 65]
                 + m1[2] * a1w1[j * 68 + 66] + m1[3] * a1w1[j * 68 + 67];
        s1 += fmaxf(u1, 0.f) * a1w2[j];
        float u2 = hproj2[b * 32 + j] + m2[0] * a2w1[j * 68 + 64] + m2[1] * a2w1[j * 68 + 65]
                 + m2[2] * a2w1[j * 68 + 66] + m2[3] * a2w1[j * 68 + 67];
        s2 += fmaxf(u2, 0.f) * a2w2[j];
      }
      sc1[i] = s1; sc2[i] = s2;
    }
    __syncthreads();
    if (tid < 16) {                              // softmax(25) + weighted sum
      const int b = tid;
      float mx = -1e30f;
      for (int n = 0; n < 25; ++n) mx = fmaxf(mx, sc1[b * 25 + n]);
      float sum = 0.f, acc0 = 0.f, acc1 = 0.f, acc2 = 0.f, acc3 = 0.f;
      for (int n = 0; n < 25; ++n) {
        const float w = expf(sc1[b * 25 + n] - mx); sum += w;
        const float* mm = &diffc[(b * 25 + n) * 4];
        acc0 += w * mm[0]; acc1 += w * mm[1]; acc2 += w * mm[2]; acc3 += w * mm[3];
      }
      attv[b * 8 + 0] = acc0 / sum; attv[b * 8 + 1] = acc1 / sum;
      attv[b * 8 + 2] = acc2 / sum; attv[b * 8 + 3] = acc3 / sum;
      mx = -1e30f;
      for (int n = 0; n < 25; ++n) mx = fmaxf(mx, sc2[b * 25 + n]);
      sum = 0.f; acc0 = acc1 = acc2 = acc3 = 0.f;
      for (int n = 0; n < 25; ++n) {
        const float w = expf(sc2[b * 25 + n] - mx); sum += w;
        const float* mm = &prevc[(b * 25 + n) * 4];
        acc0 += w * mm[0]; acc1 += w * mm[1]; acc2 += w * mm[2]; acc3 += w * mm[3];
      }
      attv[b * 8 + 4] = acc0 / sum; attv[b * 8 + 5] = acc1 / sum;
      attv[b * 8 + 6] = acc2 / sum; attv[b * 8 + 7] = acc3 / sum;
    }
    __syncthreads();
    for (int i = tid; i < 16 * 32; i += 256) {   // trans hidden (8 -> 32, relu)
      const int b = i >> 5, j = i & 31;
      float s = tb1[j];
      for (int k = 0; k < 8; ++k) s += attv[b * 8 + k] * tw1[j * 8 + k];
      ttmp[i] = fmaxf(s, 0.f);
    }
    __syncthreads();
    for (int i = tid; i < 16 * 64; i += 256) {   // x_lang = [trans_out, h_att]
      const int b = i >> 6, d = i & 63;
      float s = tb2[d];
      for (int j = 0; j < 32; ++j) s += ttmp[b * 32 + j] * tw2[d * 32 + j];
      xbuf[b * 128 + d] = s;
      xbuf[b * 128 + 64 + d] = hatt[i];
    }
    __syncthreads();
    lstm_gates(xbuf, hlang, Wil, Whl, bil, bhl, gatesL, wave, lane);
    __syncthreads();
    for (int i = tid; i < 16 * 64; i += 256) {   // lang-LSTM pointwise
      const int b = i >> 6, k = i & 63;
      const float* g = &gatesL[b * 256];
      const float ig = sigf(g[k]), fg = sigf(g[64 + k]);
      const float gg = tanhf(g[128 + k]), og = sigf(g[192 + k]);
      const float c2 = fg * clang[i] + ig * gg;
      clang[i] = c2;
      hlang[i] = og * tanhf(c2);
    }
    __syncthreads();

    // ---- vocab projection via WMMA + log-softmax ------------------------
    const int hi = lane >> 4, nn = lane & 15;
    float rmax[8];
#pragma unroll
    for (int r = 0; r < 8; ++r) rmax[r] = -1e30f;
    for (int tile = wave; tile < 126; tile += 8) {
      const int n0 = tile * 16;
      v8f acc = {};
#pragma unroll
      for (int k0 = 0; k0 < 64; k0 += 32)
        acc = WMMA_BF16(a_frag_lds(hlang, 64, k0, lane), b_frag_g(Fcw, 64, n0, k0, lane), acc);
      const int col = n0 + nn;
      const bool valid = col < 2001;
      const float bias = valid ? fcb[col] : 0.f;
#pragma unroll
      for (int r = 0; r < 8; ++r) {
        const int m = r + 8 * hi;
        const float l = acc[r] + bias;
        if (valid) {
          out[((size_t)(b0 + m) * 47 + t) * 2001 + col] = l;
          rmax[r] = fmaxf(rmax[r], l);
        }
      }
    }
#pragma unroll
    for (int r = 0; r < 8; ++r)
      for (int s = 1; s < 16; s <<= 1)
        rmax[r] = fmaxf(rmax[r], __shfl_xor(rmax[r], s, 32));
    if (nn == 0) {
#pragma unroll
      for (int r = 0; r < 8; ++r) red[wave * 16 + r + 8 * hi] = rmax[r];
    }
    __syncthreads();
    if (tid < 16) {
      float m = red[tid];
      for (int w = 1; w < 8; ++w) m = fmaxf(m, red[w * 16 + tid]);
      rowmax[tid] = m;
    }
    __syncthreads();

    float rsum[8];
#pragma unroll
    for (int r = 0; r < 8; ++r) rsum[r] = 0.f;
    for (int tile = wave; tile < 126; tile += 8) {
      const int col = tile * 16 + nn;
      if (col < 2001) {
#pragma unroll
        for (int r = 0; r < 8; ++r) {
          const int m = r + 8 * hi;
          const float l = out[((size_t)(b0 + m) * 47 + t) * 2001 + col];
          rsum[r] += expf(l - rowmax[m]);
        }
      }
    }
#pragma unroll
    for (int r = 0; r < 8; ++r)
      for (int s = 1; s < 16; s <<= 1)
        rsum[r] += __shfl_xor(rsum[r], s, 32);
    if (nn == 0) {
#pragma unroll
      for (int r = 0; r < 8; ++r) red[wave * 16 + r + 8 * hi] = rsum[r];
    }
    __syncthreads();
    if (tid < 16) {
      float s = 0.f;
      for (int w = 0; w < 8; ++w) s += red[w * 16 + tid];
      rowlz[tid] = rowmax[tid] + logf(s);
    }
    __syncthreads();
    for (int tile = wave; tile < 126; tile += 8) {
      const int col = tile * 16 + nn;
      if (col < 2001) {
#pragma unroll
        for (int r = 0; r < 8; ++r) {
          const int m = r + 8 * hi;
          const size_t idx = ((size_t)(b0 + m) * 47 + t) * 2001 + col;
          out[idx] = out[idx] - rowlz[m];
        }
      }
    }
    __syncthreads();
  }
}

extern "C" void kernel_launch(void* const* d_in, const int* in_sizes, int n_in,
                              void* d_out, int out_size, void* d_ws, size_t ws_size,
                              hipStream_t stream) {
  (void)n_in; (void)out_size; (void)ws_size;
  const int*   inst   = (const int*)d_in[0];
  const float* prevc  = (const float*)d_in[1];
  const float* finalc = (const float*)d_in[2];
  const float* embed  = (const float*)d_in[3];
  const float* wia    = (const float*)d_in[4];
  const float* wha    = (const float*)d_in[5];
  const float* bia    = (const float*)d_in[6];
  const float* bha    = (const float*)d_in[7];
  const float* wil    = (const float*)d_in[8];
  const float* whl    = (const float*)d_in[9];
  const float* bil    = (const float*)d_in[10];
  const float* bhl    = (const float*)d_in[11];
  const float* fcw    = (const float*)d_in[12];
  const float* fcb    = (const float*)d_in[13];
  const float* a1w1   = (const float*)d_in[14];
  const float* a1b1   = (const float*)d_in[15];
  const float* a1w2   = (const float*)d_in[16];
  const float* a1b2   = (const float*)d_in[17];
  const float* a2w1   = (const float*)d_in[18];
  const float* a2b1   = (const float*)d_in[19];
  const float* a2w2   = (const float*)d_in[20];
  const float* a2b2   = (const float*)d_in[21];
  const float* tw1    = (const float*)d_in[22];
  const float* tb1    = (const float*)d_in[23];
  const float* tw2    = (const float*)d_in[24];
  const float* tb2    = (const float*)d_in[25];

  bf16_t* ws = (bf16_t*)d_ws;
  const int totalw = 256 * 128 + 256 * 64 + 256 * 128 + 256 * 64 + 2016 * 64;
  cvt_weights<<<(totalw + 255) / 256, 256, 0, stream>>>(wia, wha, wil, whl, fcw, ws);

  const int Bn = in_sizes[0] / 48;               // batch
  instatt_main<<<Bn / 16, 256, 0, stream>>>(
      inst, prevc, finalc, embed, ws, bia, bha, bil, bhl, fcb,
      a1w1, a1b1, a1w2, a1b2, a2w1, a2b1, a2w2, a2b2,
      tw1, tb1, tw2, tb2, (float*)d_out);
}
